// Decoder2_27221502722677
// MI455X (gfx1250) — matrix operand order repaired and checked by reading, not compiled, not run on hardware
//
#include <hip/hip_runtime.h>

typedef __attribute__((ext_vector_type(16))) _Float16 v16h;
typedef __attribute__((ext_vector_type(8)))  float    v8f;

#define B_TOT 32768
#define T_HOR 48
#define F_DIM 16
#define H_DIM 32
#define G3H   96          // 3*H
#define TILE_M 16
#define WAVES_PER_BLOCK 8

#define LOG2E  1.44269504088896340736f

__device__ __forceinline__ float fast_sigmoid(float x) {
    // 1 / (1 + e^-x) via v_exp_f32 + v_rcp_f32 (TRANS ops, co-execute with WMMA)
    const float e = __builtin_amdgcn_exp2f(x * -LOG2E);
    return __builtin_amdgcn_rcpf(1.0f + e);
}

__device__ __forceinline__ float fast_tanh(float x) {
#if __has_builtin(__builtin_amdgcn_tanhf)
    return __builtin_amdgcn_tanhf(x);
#else
    // tanh(x) = 1 - 2/(e^(2x) + 1)
    const float e = __builtin_amdgcn_exp2f(x * (2.0f * LOG2E));
    return __builtin_fmaf(-2.0f, __builtin_amdgcn_rcpf(e + 1.0f), 1.0f);
#endif
}

__global__ __launch_bounds__(WAVES_PER_BLOCK * 32)
__attribute__((amdgpu_waves_per_eu(4)))
void Decoder2_gru_wmma_kernel(const float* __restrict__ feat,     // [B,T,F]
                              const float* __restrict__ init_in,  // [B,1]
                              const float* __restrict__ init_h,   // [B,H]
                              const float* __restrict__ Wx,       // [1+F, 3H]
                              const float* __restrict__ Wh,       // [H, 3H]
                              const float* __restrict__ bx,       // [3H]
                              const float* __restrict__ bh,       // [3H]
                              const float* __restrict__ dw,       // [H]
                              const float* __restrict__ db,       // [1]
                              float* __restrict__ out)            // [B,T]
{
    // per-wave LDS (strictly intra-wave traffic -> no block barriers needed;
    // same-wave LDS ops complete in order on CDNA5)
    __shared__ _Float16 s_h[WAVES_PER_BLOCK][TILE_M][H_DIM + 2];
    __shared__ float    s_prev[WAVES_PER_BLOCK][TILE_M];

    const int lane = threadIdx.x & 31;
    const int wave = threadIdx.x >> 5;
    const int n    = lane & 15;   // C/D: column; A: row m
    const int g    = lane >> 4;   // half (row-group for C/D, K-group for A/B)
    const int b0   = (blockIdx.x * WAVES_PER_BLOCK + wave) * TILE_M;

    // ---- B fragments (weights), f32 -> f16, resident for all 48 steps ----
    // 16-bit B 32x16 layout: element e of lane holds K = ((e&8)<<1)+(e&7)+8*g, N = n
    v16h bxf[6], bhf[6];
    #pragma unroll
    for (int tl = 0; tl < 6; ++tl) {
        #pragma unroll
        for (int e = 0; e < 16; ++e) {
            const int k = ((e & 8) << 1) + (e & 7) + 8 * g;
            const float wx = (k < 1 + F_DIM) ? Wx[k * G3H + 16 * tl + n] : 0.0f;
            bxf[tl][e] = (_Float16)wx;
            bhf[tl][e] = (_Float16)Wh[k * G3H + 16 * tl + n];
        }
    }
    // dense_w as a B fragment: only column 0 non-zero
    v16h bwf;
    #pragma unroll
    for (int e = 0; e < 16; ++e) {
        const int k = ((e & 8) << 1) + (e & 7) + 8 * g;
        bwf[e] = (n == 0) ? (_Float16)dw[k] : (_Float16)0.0f;
    }
    const float dbias = db[0];

    // biases: z/r gates get (bx+bh) fused into the first chained WMMA's C;
    // candidate tiles keep bx / bh separate (reset_after GRU needs r * (h@Wh+bh))
    float bzr[4], bxs4[2], bhs4[2];
    #pragma unroll
    for (int tl = 0; tl < 4; ++tl) bzr[tl] = bx[16 * tl + n] + bh[16 * tl + n];
    #pragma unroll
    for (int i = 0; i < 2; ++i) {
        bxs4[i] = bx[64 + 16 * i + n];
        bhs4[i] = bh[64 + 16 * i + n];
    }

    // ---- h state in C/D layout: hreg[tp][v] = h[row v+8g][col 16*tp+n] ----
    float hreg[2][8];
    #pragma unroll
    for (int tp = 0; tp < 2; ++tp)
        #pragma unroll
        for (int v = 0; v < 8; ++v)
            hreg[tp][v] = init_h[(size_t)(b0 + v + 8 * g) * H_DIM + 16 * tp + n];

    // ---- h in A layout (16x32 f16): lane row m=n, element e -> K = ((e&8)<<1)+(e&7)+8g ----
    v16h a2;
    #pragma unroll
    for (int e = 0; e < 16; ++e) {
        const int k = ((e & 8) << 1) + (e & 7) + 8 * g;
        a2[e] = (_Float16)init_h[(size_t)(b0 + n) * H_DIM + k];
    }

    if (g == 0) s_prev[wave][n] = init_in[b0 + n];

    const float4* featRow4 = (const float4*)(feat + (size_t)(b0 + n) * T_HOR * F_DIM);

    for (int t = 0; t < T_HOR; ++t) {
        const float prevf = s_prev[wave][n];

        // bulk-load this row's 16 features (4x b128), then compile-time selects
        float fv[16];
        {
            const float4 q0 = featRow4[4 * t + 0];
            const float4 q1 = featRow4[4 * t + 1];
            const float4 q2 = featRow4[4 * t + 2];
            const float4 q3 = featRow4[4 * t + 3];
            fv[0] = q0.x; fv[1] = q0.y; fv[2]  = q0.z; fv[3]  = q0.w;
            fv[4] = q1.x; fv[5] = q1.y; fv[6]  = q1.z; fv[7]  = q1.w;
            fv[8] = q2.x; fv[9] = q2.y; fv[10] = q2.z; fv[11] = q2.w;
            fv[12] = q3.x; fv[13] = q3.y; fv[14] = q3.z; fv[15] = q3.w;
        }

        // A1 = x = [prev_out | feat_t | zero-pad], 16x32 f16, A layout
        v16h a1;
        #pragma unroll
        for (int e = 0; e < 16; ++e) {
            const int kb = ((e & 8) << 1) + (e & 7);   // K for half 0
            const int k1 = kb + 8;                     // K for half 1
            const float c0 = (kb == 0) ? prevf : ((kb <= F_DIM) ? fv[kb - 1] : 0.0f);
            const float c1 = (k1 <= F_DIM) ? fv[k1 - 1] : 0.0f;
            a1[e] = (_Float16)(g ? c1 : c0);
        }

        // z/r gate tiles: chained WMMA  acc = (x@Wx + (bx+bh)) then += h@Wh
        v8f acc[4];
        #pragma unroll
        for (int tl = 0; tl < 4; ++tl) {
            v8f c;
            #pragma unroll
            for (int v = 0; v < 8; ++v) c[v] = bzr[tl];
            c = __builtin_amdgcn_wmma_f32_16x16x32_f16(
                false, a1, false, bxf[tl], (short)0, c, false, false);
            acc[tl] = __builtin_amdgcn_wmma_f32_16x16x32_f16(
                false, a2, false, bhf[tl], (short)0, c, false, false);
        }
        // candidate tiles: xh = x@Wx + bx ; hin = h@Wh + bh (kept split for r*hin)
        v8f xh[2], hin[2];
        #pragma unroll
        for (int i = 0; i < 2; ++i) {
            v8f cx, ch;
            #pragma unroll
            for (int v = 0; v < 8; ++v) { cx[v] = bxs4[i]; ch[v] = bhs4[i]; }
            xh[i]  = __builtin_amdgcn_wmma_f32_16x16x32_f16(
                false, a1, false, bxf[4 + i], (short)0, cx, false, false);
            hin[i] = __builtin_amdgcn_wmma_f32_16x16x32_f16(
                false, a2, false, bhf[4 + i], (short)0, ch, false, false);
        }

        // gates + state update in C/D layout; write new h (f16) to LDS for transpose
        #pragma unroll
        for (int tp = 0; tp < 2; ++tp) {
            #pragma unroll
            for (int v = 0; v < 8; ++v) {
                const float z  = fast_sigmoid(acc[tp][v]);
                const float r  = fast_sigmoid(acc[2 + tp][v]);
                const float hh = fast_tanh(__builtin_fmaf(r, hin[tp][v], xh[tp][v]));
                const float hn = z * hreg[tp][v] + (1.0f - z) * hh;
                hreg[tp][v] = hn;
                s_h[wave][v + 8 * g][16 * tp + n] = (_Float16)hn;
            }
        }

        // reload new h in A layout (carried into next step's GEMM2)
        #pragma unroll
        for (int e = 0; e < 16; ++e) {
            const int k = ((e & 8) << 1) + (e & 7) + 8 * g;
            a2[e] = s_h[wave][n][k];
        }

        // Dense(1) as a 13th WMMA: out column 0 = h_new @ dense_w + dense_b
        v8f co;
        #pragma unroll
        for (int v = 0; v < 8; ++v) co[v] = (n == 0) ? dbias : 0.0f;
        const v8f acco = __builtin_amdgcn_wmma_f32_16x16x32_f16(
            false, a2, false, bwf, (short)0, co, false, false);

        if (n == 0) {
            #pragma unroll
            for (int v = 0; v < 8; ++v) {
                const int row = b0 + v + 8 * g;
                const float o = acco[v];
                out[(size_t)row * T_HOR + t] = o;
                s_prev[wave][v + 8 * g] = o;   // prev_out for next step
            }
        }
    }
}

extern "C" void kernel_launch(void* const* d_in, const int* in_sizes, int n_in,
                              void* d_out, int out_size, void* d_ws, size_t ws_size,
                              hipStream_t stream) {
    const float* feat    = (const float*)d_in[0];  // decoder_feature [B,T,F]
    const float* init_in = (const float*)d_in[1];  // decoder_init_input [B,1]
    const float* init_h  = (const float*)d_in[2];  // init_state [B,H]
    const float* Wx      = (const float*)d_in[3];  // kernel [17, 96]
    const float* Wh      = (const float*)d_in[4];  // recurrent_kernel [32, 96]
    const float* bx      = (const float*)d_in[5];  // bias_x [96]
    const float* bh      = (const float*)d_in[6];  // bias_h [96]
    const float* dw      = (const float*)d_in[7];  // dense_w [32]
    const float* db      = (const float*)d_in[8];  // dense_b [1]
    float* outp          = (float*)d_out;          // [B,T,1]

    dim3 block(WAVES_PER_BLOCK * 32);                 // 256 threads = 8 waves
    dim3 grid(B_TOT / (TILE_M * WAVES_PER_BLOCK));    // 2048 tiles / 8 = 256 blocks
    Decoder2_gru_wmma_kernel<<<grid, block, 0, stream>>>(
        feat, init_in, init_h, Wx, Wh, bx, bh, dw, db, outp);
}
